// DialogueSNN_14877766713831
// MI455X (gfx1250) — matrix operand order repaired and checked by reading, not compiled, not run on hardware
//
#include <hip/hip_runtime.h>
#include <hip/hip_bf16.h>

// ---------------------------------------------------------------------------
// DialogueSNN on MI455X (gfx1250, wave32, WMMA).
//
// Problem: B=8, S=256, V=32000, H=128, E=64, 20 LIF steps/token.
//   5120 sequential global steps; per step cur2 = spk1 @ W2 + b2 (8x128 @ 128x32000).
// Plan:
//   K0: pack W2 (f32) -> f16 in WMMA B-fragment order (one contiguous v16h/lane).
//   K1: sequential mem1 trajectory (tiny state: 8x128). Emits spk1 for all 5120
//       steps as f16 row-major [step*8+b][128]; writes final mem1 to d_out.
//   K2: fused GEMM+LIF. One wave per 16-wide v-tile (2000 waves). W2 fragments
//       pinned in 32 VGPRs, mem2 in 8 VGPRs. Loop over 2560 2-step tiles:
//       4x v_wmma_f32_16x16x32_f16 (M=16 = 2 steps x 8 batch), then LIF update
//       (shfl_xor to serialize the 2 steps). Avoids the 10.5 GB cur2 round-trip
//       a phase-split design would pay (~0.45 ms at 23.3 TB/s); HBM traffic is
//       just the 263 MB output + ~18 MB workspace.
// ---------------------------------------------------------------------------

typedef __attribute__((ext_vector_type(16))) _Float16 v16h;
typedef __attribute__((ext_vector_type(8)))  _Float16 v8h;
typedef __attribute__((ext_vector_type(8)))  float    v8f;

#define SNN_B   8
#define SNN_S   256
#define SNN_V   32000
#define SNN_H   128
#define SNN_E   64
#define SNN_T   20
#define GSTEPS  (SNN_S * SNN_T)          // 5120 global steps
#define NTILES  (GSTEPS / 2)             // 2560 two-step row tiles
#define VTILES  (SNN_V / 16)             // 2000 column tiles
#define BETA    0.95f
#define THRESH  1.0f

// ---------------------------------------------------------------------------
// K0: pack W2 [128 x 32000] f32 -> f16 WMMA-B fragment layout.
// Bbuf index = (((vt*4 + kt)*32 + lane)*16 + e)
//   column n = lane & 15 ; K = kt*32 + ((lane>>4)<<4) + e
// so each lane of K2 loads its fragment as ONE contiguous 32B v16h.
// ---------------------------------------------------------------------------
__global__ void __launch_bounds__(256) snn_pack_w2(const float* __restrict__ W2,
                                                   _Float16* __restrict__ Bbuf) {
    int idx  = blockIdx.x * 256 + threadIdx.x;      // grid sized exactly
    int e    = idx & 15;
    int lane = (idx >> 4) & 31;
    int kt   = (idx >> 9) & 3;
    int vt   = idx >> 11;
    int n    = lane & 15;
    int K    = kt * 32 + ((lane >> 4) << 4) + e;
    int v    = vt * 16 + n;
    Bbuf[idx] = (_Float16)W2[K * SNN_V + v];
}

// ---------------------------------------------------------------------------
// K1: sequential LIF-1 trajectory. One workgroup, 1024 threads = (b,h).
// cur1[b,h] = embed[x[b,t]] . W1[:,h] + b1[h]  (W1 staged in LDS, 32 KB).
// Emits spk1 as f16, row-major rows r = (t*20+s)*8 + b, 128 cols.
// ---------------------------------------------------------------------------
__global__ void __launch_bounds__(1024) snn_lif1(const int*   __restrict__ x,
                                                 const float* __restrict__ embed,
                                                 const float* __restrict__ W1,
                                                 const float* __restrict__ b1,
                                                 _Float16*    __restrict__ spk1,
                                                 float*       __restrict__ mem1_out) {
    __shared__ float w1s[SNN_E * SNN_H];            // 32 KB
    for (int i = threadIdx.x; i < SNN_E * SNN_H; i += 1024) w1s[i] = W1[i];
    __syncthreads();

    const int b = threadIdx.x >> 7;                 // 0..7
    const int h = threadIdx.x & 127;                // 0..127
    const float b1v = b1[h];
    float m = 0.0f;

    for (int t = 0; t < SNN_S; ++t) {
        const int token = x[b * SNN_S + t];
        const float* er = embed + (long long)token * SNN_E;
        float cur = b1v;
        #pragma unroll 8
        for (int e = 0; e < SNN_E; ++e) cur = fmaf(er[e], w1s[e * SNN_H + h], cur);

        #pragma unroll
        for (int s = 0; s < SNN_T; ++s) {
            float r = (m > THRESH) ? 1.0f : 0.0f;   // reset from OLD mem
            m = fmaf(BETA, m, cur - r);             // mem = beta*mem + cur - reset
            _Float16 spk = (m > THRESH) ? (_Float16)1.0f : (_Float16)0.0f;
            spk1[(long long)((t * SNN_T + s) * SNN_B + b) * SNN_H + h] = spk;
        }
    }
    mem1_out[b * SNN_H + h] = m;
}

// ---------------------------------------------------------------------------
// K2: fused spk1@W2 + LIF-2 scan. 1 wave per v-tile; 4 waves / block.
// M-tile = 16 rows = steps (2i, 2i+1) x batch 0..7. C/D layout:
//   VGPR i, lanes 0-15 : (step 2i,   b=i, col=lane)
//   VGPR i, lanes 16-31: (step 2i+1, b=i, col=lane-16)
// mem2 kept canonically in lanes 0-15; shfl_xor(16) fetches step-b currents.
// ---------------------------------------------------------------------------
__global__ void __launch_bounds__(128) snn_fused(const _Float16* __restrict__ spk1,
                                                 const _Float16* __restrict__ Bbuf,
                                                 const float*    __restrict__ b2,
                                                 float*          __restrict__ out,       // [B,S,V]
                                                 float*          __restrict__ mem2_out)  // [B,V]
{
    const int lane = threadIdx.x & 31;
    const int vt   = blockIdx.x * 4 + (threadIdx.x >> 5);   // 0..1999 exactly

    // Pin this wave's W2 fragments (16 cols x K=128, f16) in 32 VGPRs.
    v16h bf[4];
    #pragma unroll
    for (int kt = 0; kt < 4; ++kt)
        bf[kt] = *(const v16h*)(Bbuf + (((long long)vt * 4 + kt) * 32 + lane) * 16);

    const float b2c = b2[vt * 16 + (lane & 15)];

    v8f m;                                          // mem2 for b=0..7 (lanes 0-15 valid)
    #pragma unroll
    for (int i = 0; i < 8; ++i) m[i] = 0.0f;

    // A-fragment per-lane base: row (lane&15), K half-select by lane>=16.
    const int aoff = (lane & 15) * SNN_H + ((lane >> 4) << 4);

    for (int tile = 0; tile < NTILES; ++tile) {
        const _Float16* abase = spk1 + (long long)tile * 16 * SNN_H;
        __builtin_prefetch(abase + 16 * SNN_H, 0, 1);   // next tile -> global_prefetch

        v8f c = {};                                  // fresh cur2 accumulator
        #pragma unroll
        for (int kt = 0; kt < 4; ++kt) {
            v16h a;
            const _Float16* p = abase + aoff + kt * 32;
            ((v8h*)&a)[0] = *(const v8h*)(p);        // K 0..7   (or 16..23)
            ((v8h*)&a)[1] = *(const v8h*)(p + 8);    // K 8..15  (or 24..31)
            c = __builtin_amdgcn_wmma_f32_16x16x32_f16(
                    false, a, false, bf[kt], (short)0, c, false, false);
        }

        // LIF-2: two sequential steps. Lanes 0-15 own state; pull step-b lanes.
        #pragma unroll
        for (int i = 0; i < 8; ++i) {
            float ca = c[i];
            float cb = __shfl_xor(c[i], 16, 32);     // step 2*tile+1 current
            float r0 = (m[i] > THRESH) ? 1.0f : 0.0f;
            m[i] = fmaf(BETA, m[i], ca + b2c - r0);
            float r1 = (m[i] > THRESH) ? 1.0f : 0.0f;
            m[i] = fmaf(BETA, m[i], cb + b2c - r1);
        }

        // Second step of every 10th tile is inner step 19 -> emit spk2.
        if (((tile % 10) == 9) && lane < 16) {
            const int t   = tile / 10;
            const int col = vt * 16 + lane;
            #pragma unroll
            for (int i = 0; i < 8; ++i)
                out[((long long)i * SNN_S + t) * SNN_V + col] =
                    (m[i] > THRESH) ? 1.0f : 0.0f;
        }
    }

    if (lane < 16) {
        const int col = vt * 16 + lane;
        #pragma unroll
        for (int i = 0; i < 8; ++i)
            mem2_out[(long long)i * SNN_V + col] = m[i];
    }
}

// ---------------------------------------------------------------------------
extern "C" void kernel_launch(void* const* d_in, const int* in_sizes, int n_in,
                              void* d_out, int out_size, void* d_ws, size_t ws_size,
                              hipStream_t stream) {
    const int*   x     = (const int*)  d_in[0];   // [8,256]
    const float* embed = (const float*)d_in[1];   // [32000,64]
    const float* W1    = (const float*)d_in[2];   // [64,128]
    const float* b1    = (const float*)d_in[3];   // [128]
    const float* W2    = (const float*)d_in[4];   // [128,32000]
    const float* b2    = (const float*)d_in[5];   // [32000]

    float* out      = (float*)d_out;                          // [8,256,32000]
    float* mem1_out = out + (long long)SNN_B * SNN_S * SNN_V; // [8,128]
    float* mem2_out = mem1_out + SNN_B * SNN_H;               // [8,32000]

    char* ws = (char*)d_ws;
    _Float16* spk1 = (_Float16*)ws;                           // 5120*8*128*2 = 10,485,760 B
    _Float16* Bbuf = (_Float16*)(ws + 10485760);              // 2000*4*32*16*2 = 8,192,000 B

    // K0: pack W2 -> f16 B-fragments (4,096,000 elements, exact grid).
    snn_pack_w2<<<16000, 256, 0, stream>>>(W2, Bbuf);
    // K1: sequential mem1/spk1 trajectory (single workgroup).
    snn_lif1<<<1, 1024, 0, stream>>>(x, embed, W1, b1, spk1, mem1_out);
    // K2: fused WMMA GEMM + mem2 LIF scan: 2000 waves = 500 blocks x 4 waves.
    snn_fused<<<VTILES / 4, 128, 0, stream>>>(spk1, Bbuf, b2, out, mem2_out);
}
